// Attention_23905787969932
// MI455X (gfx1250) — matrix operand order repaired and checked by reading, not compiled, not run on hardware
//
#include <hip/hip_runtime.h>
#include <hip/hip_bf16.h>

// ---------------------------------------------------------------------------
// GQA attention block for MI455X (gfx1250, wave32, WMMA).
// v2.1: fix async-to-LDS builtin argument types (int4* per hipcc diagnostic).
// DPP row reductions, LDS staging of K/V and GEMM-B tiles, async global->LDS.
// ---------------------------------------------------------------------------

typedef __attribute__((ext_vector_type(16))) __bf16 bf16x16;
typedef __attribute__((ext_vector_type(8)))  float  f32x8;
typedef __attribute__((vector_size(16)))     int    i32x4_t;

union Frag16 {
  bf16x16 v;
  unsigned short u[16];
  unsigned int   w[8];
};

#define HIDDEN_DIM 2048
#define SEQ        2048
#define BATCH      2
#define NHEADS     16
#define HEADDIM    128
#define KVDIM      512      // n_kv_heads * head_dim
#define ROWS       4096     // BATCH * SEQ

#define AS1 __attribute__((address_space(1)))
#define AS3 __attribute__((address_space(3)))

#if defined(__has_builtin)
#if __has_builtin(__builtin_amdgcn_global_load_async_to_lds_b128)
#define ASYNC_LDS 1
#endif
#endif
#ifndef ASYNC_LDS
#define ASYNC_LDS 0
#endif

__device__ __forceinline__ unsigned short f2bf(float f) {
  unsigned int u = __float_as_uint(f);
  u += 0x7FFFu + ((u >> 16) & 1u);   // round-to-nearest-even
  return (unsigned short)(u >> 16);
}

// 16-lane row allreduce via DPP rotations (VALU only, no LDS round trips).
// row_ror:N ctrl = 0x120 | N ; rows of 16 lanes are independent.
__device__ __forceinline__ float row16_max(float x) {
  x = fmaxf(x, __uint_as_float(__builtin_amdgcn_mov_dpp(__float_as_uint(x), 0x121, 0xf, 0xf, false)));
  x = fmaxf(x, __uint_as_float(__builtin_amdgcn_mov_dpp(__float_as_uint(x), 0x122, 0xf, 0xf, false)));
  x = fmaxf(x, __uint_as_float(__builtin_amdgcn_mov_dpp(__float_as_uint(x), 0x124, 0xf, 0xf, false)));
  x = fmaxf(x, __uint_as_float(__builtin_amdgcn_mov_dpp(__float_as_uint(x), 0x128, 0xf, 0xf, false)));
  return x;
}
__device__ __forceinline__ float row16_sum(float x) {
  x += __uint_as_float(__builtin_amdgcn_mov_dpp(__float_as_uint(x), 0x121, 0xf, 0xf, false));
  x += __uint_as_float(__builtin_amdgcn_mov_dpp(__float_as_uint(x), 0x122, 0xf, 0xf, false));
  x += __uint_as_float(__builtin_amdgcn_mov_dpp(__float_as_uint(x), 0x124, 0xf, 0xf, false));
  x += __uint_as_float(__builtin_amdgcn_mov_dpp(__float_as_uint(x), 0x128, 0xf, 0xf, false));
  return x;
}

// Load 32 contiguous bytes into a fragment (two ds/global b128 loads).
__device__ __forceinline__ void ld32B(Frag16& f, const unsigned short* p) {
  const uint4* q = (const uint4*)p;
  uint4 a = q[0], b = q[1];
  f.w[0] = a.x; f.w[1] = a.y; f.w[2] = a.z; f.w[3] = a.w;
  f.w[4] = b.x; f.w[5] = b.y; f.w[6] = b.z; f.w[7] = b.w;
}

// --------------------------- fp32 -> bf16 convert ---------------------------
__global__ __launch_bounds__(256) void cvt_f32_bf16(const float* __restrict__ in,
                                                    unsigned short* __restrict__ out,
                                                    int n) {
  int i = blockIdx.x * blockDim.x + threadIdx.x;
  if (i < n) out[i] = f2bf(in[i]);
}

// ------------------------------- bf16 GEMM ----------------------------------
// C[M,N] = A[M,K] * B[K,N]. One wave computes a 16x64 tile; the 4 waves of a
// block share the same 64 B-columns, so B k-slices are staged (transposed)
// in LDS once per block and read back as contiguous ds_load_b128 fragments.
template <bool OUT_BF16>
__global__ __launch_bounds__(128) void gemm_bf16(
    const unsigned short* __restrict__ A,
    const unsigned short* __restrict__ B,
    float* __restrict__ Cf,
    unsigned short* __restrict__ Cb,
    int M, int N, int Kdim) {
  __shared__ unsigned short sBt[64][32];   // [n-local][k-local], 4 KiB

  const int tid  = threadIdx.x;
  const int wave = tid >> 5;
  const int lane = tid & 31;
  const int half = lane >> 4;
  const int lcol = lane & 15;
  const int m0 = (blockIdx.y * 4 + wave) * 16;
  const int n0 = blockIdx.x * 64;

  f32x8 acc[4];
#pragma unroll
  for (int t = 0; t < 4; ++t) acc[t] = (f32x8)0.0f;

  const size_t arow = (size_t)(m0 + lcol) * (size_t)Kdim;
  for (int k0 = 0; k0 < Kdim; k0 += 32) {
    // ---- stage B[k0..k0+31][n0..n0+63] transposed into LDS ----
    __syncthreads();
#pragma unroll
    for (int c = 0; c < 2; ++c) {
      int q = tid + c * 128;              // 256 chunks of 8 shorts
      int krow = q >> 3;
      int ncol = (q & 7) << 3;
      uint4 d = *(const uint4*)(B + (size_t)(k0 + krow) * (size_t)N + n0 + ncol);
      const unsigned short* dp = (const unsigned short*)&d;
#pragma unroll
      for (int j = 0; j < 8; ++j) sBt[ncol + j][krow] = dp[j];
    }
    __syncthreads();

    // ---- A fragment (16x32): packed 16B global loads ----
    Frag16 a;
#pragma unroll
    for (int i = 0; i < 8; ++i) {
      int kk = k0 + (half << 4) + (i << 1);
      a.w[i] = *(const unsigned int*)(A + arow + kk);
    }
#pragma unroll
    for (int t = 0; t < 4; ++t) {
      Frag16 b;                           // B frag: N=lane%16, K=half*16+e
      ld32B(b, &sBt[t * 16 + lcol][half * 16]);
      acc[t] = __builtin_amdgcn_wmma_f32_16x16x32_bf16(
          false, a.v, false, b.v, (short)0, acc[t], false, false);
    }
  }
#pragma unroll
  for (int t = 0; t < 4; ++t) {
#pragma unroll
    for (int r = 0; r < 8; ++r) {
      int row = m0 + r + (half << 3);
      int col = n0 + t * 16 + lcol;
      if (OUT_BF16) Cb[(size_t)row * (size_t)N + col] = f2bf(acc[t][r]);
      else          Cf[(size_t)row * (size_t)N + col] = acc[t][r];
    }
  }
}

// --------------------------- flash attention (GQA) --------------------------
// Block = 4 waves x 16 queries, one (batch, head). All waves share the same
// K/V: K tile staged row-major in LDS (async global->LDS when available),
// V tile staged transposed so P*V B-fragments are contiguous ds_load_b128.
__global__ __launch_bounds__(128) void attn_fa(
    const unsigned short* __restrict__ Q,   // [4096][2048] bf16 bits
    const unsigned short* __restrict__ K,   // [4096][512]
    const unsigned short* __restrict__ V,   // [4096][512]
    unsigned short* __restrict__ O) {       // [4096][2048]
  __shared__ unsigned short sK[32][128];    // keys x d, 8 KiB
  __shared__ unsigned short sVt[128][32];   // d x keys, 8 KiB
  __shared__ unsigned short pst[4][16][34]; // per-wave P staging (padded)

  const int tid  = threadIdx.x;
  const int wave = tid >> 5;
  const int lane = tid & 31;
  const int half = lane >> 4;
  const int lcol = lane & 15;
  const int q0 = blockIdx.x * 64 + wave * 16;  // seq position within batch
  const int h  = blockIdx.y;
  const int b  = blockIdx.z;
  const float scale = 0.08838834764831845f;    // 1/sqrt(128)

  const unsigned short* Qb = Q + (size_t)b * SEQ * HIDDEN_DIM + (size_t)h * HEADDIM;
  const unsigned short* Kb = K + (size_t)b * SEQ * KVDIM + (size_t)(h >> 2) * HEADDIM;
  const unsigned short* Vb = V + (size_t)b * SEQ * KVDIM + (size_t)(h >> 2) * HEADDIM;

  // Q A-fragments for all 4 d-chunks of 32, loaded once (global b128s).
  Frag16 qa[4];
  const size_t qrow = (size_t)(q0 + lcol) * HIDDEN_DIM;
#pragma unroll
  for (int c = 0; c < 4; ++c)
#pragma unroll
    for (int i = 0; i < 8; ++i) {
      int kk = c * 32 + (half << 4) + (i << 1);
      qa[c].w[i] = *(const unsigned int*)(Qb + qrow + kk);
    }

  f32x8 o[8];
  float m[8], l[8];
#pragma unroll
  for (int j = 0; j < 8; ++j) o[j] = (f32x8)0.0f;
#pragma unroll
  for (int r = 0; r < 8; ++r) { m[r] = -1e30f; l[r] = 0.0f; }

  for (int kb = 0; kb < SEQ; kb += 32) {
    // ---- cooperative staging of K (row-major) and V (transposed) ----
    __syncthreads();  // readers of previous tile are done
#pragma unroll
    for (int c = 0; c < 4; ++c) {
      int q = tid + c * 128;               // 512 chunks of 8 shorts per tile
      int key  = q >> 4;
      int dcol = (q & 15) << 3;
      const unsigned short* gK = Kb + (size_t)(kb + key) * KVDIM + dcol;
#if ASYNC_LDS
      __builtin_amdgcn_global_load_async_to_lds_b128(
          (AS1 i32x4_t*)(uintptr_t)gK,
          (AS3 i32x4_t*)(unsigned int)(uintptr_t)(&sK[key][dcol]), 0, 0);
#else
      *(uint4*)&sK[key][dcol] = *(const uint4*)gK;
#endif
      uint4 dv = *(const uint4*)(Vb + (size_t)(kb + key) * KVDIM + dcol);
      const unsigned short* dp = (const unsigned short*)&dv;
#pragma unroll
      for (int j = 0; j < 8; ++j) sVt[dcol + j][key] = dp[j];
    }
#if ASYNC_LDS
#if __has_builtin(__builtin_amdgcn_s_wait_asynccnt)
    __builtin_amdgcn_s_wait_asynccnt(0);
#else
    asm volatile("s_wait_asynccnt 0x0" ::: "memory");
#endif
#endif
    __syncthreads();

    // ---- scores: two 16x16 tiles (keys kb..+15, kb+16..+31) ----
    f32x8 sc[2];
#pragma unroll
    for (int t = 0; t < 2; ++t) {
      f32x8 c = (f32x8)0.0f;
#pragma unroll
      for (int c4 = 0; c4 < 4; ++c4) {
        Frag16 kf;   // B = K^T frag: N = key (lane), K = d (contiguous)
        ld32B(kf, &sK[t * 16 + lcol][c4 * 32 + half * 16]);
        c = __builtin_amdgcn_wmma_f32_16x16x32_bf16(
            false, qa[c4].v, false, kf.v, (short)0, c, false, false);
      }
      sc[t] = c;
    }

    // ---- online softmax: element r -> row = r + 8*half ----
    float corr[8];
#pragma unroll
    for (int r = 0; r < 8; ++r) {
      float s0 = sc[0][r] * scale;
      float s1 = sc[1][r] * scale;
      float mn = fmaxf(m[r], row16_max(fmaxf(s0, s1)));
      float p0 = __expf(s0 - mn);
      float p1 = __expf(s1 - mn);
      float ls = row16_sum(p0 + p1);
      float cr = __expf(m[r] - mn);
      l[r] = l[r] * cr + ls;
      m[r] = mn;
      corr[r] = cr;
      sc[0][r] = p0;
      sc[1][r] = p1;
    }
#pragma unroll
    for (int j = 0; j < 8; ++j)
#pragma unroll
      for (int r = 0; r < 8; ++r) o[j][r] *= corr[r];

    // ---- restage P (C layout) as an A fragment via per-wave LDS ----
    __syncthreads();
#pragma unroll
    for (int r = 0; r < 8; ++r) {
      pst[wave][r + (half << 3)][lcol]      = f2bf(sc[0][r]);
      pst[wave][r + (half << 3)][16 + lcol] = f2bf(sc[1][r]);
    }
    __syncthreads();
    Frag16 pa;
#pragma unroll
    for (int i = 0; i < 8; ++i) {
      int kk = (half << 4) + (i << 1);
      pa.w[i] = *(const unsigned int*)&pst[wave][lcol][kk];
    }

    // ---- P(16x32) x V(32x128): 8 WMMA, V frags contiguous from sVt ----
#pragma unroll
    for (int j = 0; j < 8; ++j) {
      Frag16 vf;   // element e -> key = half*16+e, col = j*16+lcol
      ld32B(vf, &sVt[j * 16 + lcol][half * 16]);
      o[j] = __builtin_amdgcn_wmma_f32_16x16x32_bf16(
          false, pa.v, false, vf.v, (short)0, o[j], false, false);
    }
  }

  // ---- normalize by row sum and store [b, s, h*128 + d] as bf16 ----
#pragma unroll
  for (int j = 0; j < 8; ++j) {
#pragma unroll
    for (int r = 0; r < 8; ++r) {
      int row = q0 + r + (half << 3);
      int col = h * HEADDIM + j * 16 + lcol;
      O[((size_t)b * SEQ + row) * HIDDEN_DIM + col] = f2bf(o[j][r] / l[r]);
    }
  }
}

// ------------------------------- launcher -----------------------------------
extern "C" void kernel_launch(void* const* d_in, const int* in_sizes, int n_in,
                              void* d_out, int out_size, void* d_ws, size_t ws_size,
                              hipStream_t stream) {
  const float* hidden = (const float*)d_in[0];
  // d_in[1] = position_ids: unused (reference applies no RoPE)
  const float* w_q = (const float*)d_in[2];
  const float* w_k = (const float*)d_in[3];
  const float* w_v = (const float*)d_in[4];
  const float* w_o = (const float*)d_in[5];
  float* out = (float*)d_out;

  char* ws = (char*)d_ws;
  unsigned short* X  = (unsigned short*)(ws);              // 16 MiB
  unsigned short* Wq = (unsigned short*)(ws + 16777216);   //  8 MiB
  unsigned short* Wk = (unsigned short*)(ws + 25165824);   //  2 MiB
  unsigned short* Wv = (unsigned short*)(ws + 27262976);   //  2 MiB
  unsigned short* Wo = (unsigned short*)(ws + 29360128);   //  8 MiB
  unsigned short* Qm = (unsigned short*)(ws + 37748736);   // 16 MiB
  unsigned short* Km = (unsigned short*)(ws + 54525952);   //  4 MiB
  unsigned short* Vm = (unsigned short*)(ws + 58720256);   //  4 MiB
  unsigned short* Am = (unsigned short*)(ws + 62914560);   // 16 MiB (~76 MiB)

  {
    int n;
    n = ROWS * HIDDEN_DIM;
    cvt_f32_bf16<<<(n + 255) / 256, 256, 0, stream>>>(hidden, X, n);
    n = HIDDEN_DIM * HIDDEN_DIM;
    cvt_f32_bf16<<<(n + 255) / 256, 256, 0, stream>>>(w_q, Wq, n);
    n = HIDDEN_DIM * KVDIM;
    cvt_f32_bf16<<<(n + 255) / 256, 256, 0, stream>>>(w_k, Wk, n);
    cvt_f32_bf16<<<(n + 255) / 256, 256, 0, stream>>>(w_v, Wv, n);
    n = HIDDEN_DIM * HIDDEN_DIM;
    cvt_f32_bf16<<<(n + 255) / 256, 256, 0, stream>>>(w_o, Wo, n);
  }

  dim3 blk(128);
  gemm_bf16<true><<<dim3(HIDDEN_DIM / 64, ROWS / 64), blk, 0, stream>>>(
      X, Wq, nullptr, Qm, ROWS, HIDDEN_DIM, HIDDEN_DIM);
  gemm_bf16<true><<<dim3(KVDIM / 64, ROWS / 64), blk, 0, stream>>>(
      X, Wk, nullptr, Km, ROWS, KVDIM, HIDDEN_DIM);
  gemm_bf16<true><<<dim3(KVDIM / 64, ROWS / 64), blk, 0, stream>>>(
      X, Wv, nullptr, Vm, ROWS, KVDIM, HIDDEN_DIM);

  attn_fa<<<dim3(SEQ / 64, NHEADS, BATCH), blk, 0, stream>>>(Qm, Km, Vm, Am);

  gemm_bf16<false><<<dim3(HIDDEN_DIM / 64, ROWS / 64), blk, 0, stream>>>(
      Am, Wo, out, nullptr, ROWS, HIDDEN_DIM, HIDDEN_DIM);
}
